// PairedAug_34952443855245
// MI455X (gfx1250) — compile-verified, matrix operand order checked by reading, not run on hardware
//
#include <hip/hip_runtime.h>
#include <math.h>

typedef __attribute__((ext_vector_type(16))) _Float16 v16h;
typedef __attribute__((ext_vector_type(8)))  float    v8f;

#define Bn  32
#define Hn  512
#define Wn  512
#define HWn (Hn * Wn)

__device__ __forceinline__ float clamp01(float x) {
    return fminf(fmaxf(x, 0.0f), 1.0f);
}

// Kernel 0: zero the per-image gray accumulators (ws is poisoned by the harness).
__global__ void PairedAug_zero_kernel(float* __restrict__ gsum) {
    if (threadIdx.x < Bn) gsum[threadIdx.x] = 0.0f;
}

// Kernel A: fused flips + rotate + affine (exact 16-tap double-bilinear) + brightness,
// writes imgs (brightness-clipped) and masks to d_out, and reduces per-image gray sums
// on the matrix pipe (v_wmma_f32_16x16x32_f16 against an all-ones B tile).
__global__ __launch_bounds__(256) void PairedAug_geom_kernel(
    const float* __restrict__ imgs,  const float* __restrict__ masks,
    const int*   __restrict__ hflip, const int*   __restrict__ vflip,
    const float* __restrict__ angles, const float* __restrict__ translate,
    const float* __restrict__ scale,  const float* __restrict__ bright,
    float* __restrict__ out_imgs, float* __restrict__ out_masks,
    float* __restrict__ gsum)
{
    const int b    = (int)blockIdx.x >> 6;   // 64 blocks per image
    const int tile = (int)blockIdx.x & 63;
    const int tid  = (int)threadIdx.x;

    const float th    = angles[b] * (3.14159265358979323846f / 180.0f);
    const float cth   = cosf(th);
    const float sth   = sinf(th);
    const float tx    = translate[2 * b + 0] * (float)Wn;
    const float ty    = translate[2 * b + 1] * (float)Hn;
    const float invsc = 1.0f / scale[b];
    const float bf    = bright[b];
    const bool  hfl   = hflip[b] > 0;
    const bool  vfl   = vflip[b] > 0;
    const float cx    = (float)(Wn - 1) * 0.5f;
    const float cy    = (float)(Hn - 1) * 0.5f;
    const float lim   = (float)(Wn - 1);

    const float* pR = imgs  + (size_t)(b * 3 + 0) * HWn;
    const float* pG = imgs  + (size_t)(b * 3 + 1) * HWn;
    const float* pB = imgs  + (size_t)(b * 3 + 2) * HWn;
    const float* pM = masks + (size_t)b * HWn;
    float* oR = out_imgs  + (size_t)(b * 3 + 0) * HWn;
    float* oG = out_imgs  + (size_t)(b * 3 + 1) * HWn;
    float* oB = out_imgs  + (size_t)(b * 3 + 2) * HWn;
    float* oM = out_masks + (size_t)b * HWn;

    const int base = tile * 4096 + tid;

    v16h grayA = {};

#pragma unroll
    for (int k = 0; k < 16; ++k) {
        const int p = base + (k << 8);
        const int y = p >> 9;
        const int x = p & (Wn - 1);

        // ---- affine (inverse map): source coords into the rotated image ----
        const float ax  = ((float)x - cx - tx) * invsc + cx;
        const float ay  = ((float)y - cy - ty) * invsc + cy;
        const float ax0 = floorf(ax);
        const float ay0 = floorf(ay);
        const float wax = ax - ax0;
        const float way = ay - ay0;

        float accR = 0.0f, accG = 0.0f, accB = 0.0f, accM = 0.0f;

#pragma unroll
        for (int j = 0; j < 2; ++j) {
#pragma unroll
            for (int i = 0; i < 2; ++i) {
                const float xf = ax0 + (float)i;
                const float yf = ay0 + (float)j;
                const float wa = (i ? wax : 1.0f - wax) * (j ? way : 1.0f - way);
                const float va = (xf >= 0.0f && xf <= lim && yf >= 0.0f && yf <= lim) ? 1.0f : 0.0f;
                const float w  = wa * va;

                // rotated-image pixel value at integer coords (clipped like the reference)
                const float xc = fminf(fmaxf(xf, 0.0f), lim);
                const float yc = fminf(fmaxf(yf, 0.0f), lim);
                const float dx = xc - cx;
                const float dy = yc - cy;
                float rx =  cth * dx + sth * dy + cx;
                float ry = -sth * dx + cth * dy + cy;
                // flips applied to the image BEFORE rotation == reflect the sample coords
                rx = hfl ? (lim - rx) : rx;
                ry = vfl ? (lim - ry) : ry;

                const float rx0 = floorf(rx);
                const float ry0 = floorf(ry);
                const float wrx = rx - rx0;
                const float wry = ry - ry0;

                float rr = 0.0f, rg = 0.0f, rb = 0.0f, rm = 0.0f;
#pragma unroll
                for (int jj = 0; jj < 2; ++jj) {
#pragma unroll
                    for (int ii = 0; ii < 2; ++ii) {
                        const float xrf = rx0 + (float)ii;
                        const float yrf = ry0 + (float)jj;
                        const float wr  = (ii ? wrx : 1.0f - wrx) * (jj ? wry : 1.0f - wry);
                        const float vr  = (xrf >= 0.0f && xrf <= lim &&
                                           yrf >= 0.0f && yrf <= lim) ? 1.0f : 0.0f;
                        const int xi  = (int)fminf(fmaxf(xrf, 0.0f), lim);
                        const int yi  = (int)fminf(fmaxf(yrf, 0.0f), lim);
                        const int off = yi * Wn + xi;
                        const float wv = wr * vr;
                        rr = fmaf(wv, pR[off], rr);
                        rg = fmaf(wv, pG[off], rg);
                        rb = fmaf(wv, pB[off], rb);
                        rm = fmaf(wv, pM[off], rm);
                    }
                }
                accR = fmaf(w, rr, accR);
                accG = fmaf(w, rg, accG);
                accB = fmaf(w, rb, accB);
                accM = fmaf(w, rm, accM);
            }
        }

        // ---- brightness (clipped) + write-out ----
        const float r1 = clamp01(accR * bf);
        const float g1 = clamp01(accG * bf);
        const float b1 = clamp01(accB * bf);
        oR[p] = r1;
        oG[p] = g1;
        oB[p] = b1;
        oM[p] = accM;

        grayA[k] = (_Float16)(0.299f * r1 + 0.587f * g1 + 0.114f * b1);
    }

    // ---- wave-level gray reduction on the matrix pipe ----
    // A = 16x32 f16 tile holding 512 gray values (16 per lane), B = all-ones.
    // D[m,n] = rowsum_m with f32 accumulation; lanes 0-15 hold rows 0-7 in their
    // 8 accumulators, lanes 16-31 hold rows 8-15 -> local sum + shfl_xor(16).
    v16h ones;
#pragma unroll
    for (int i = 0; i < 16; ++i) ones[i] = (_Float16)1.0f;

    v8f c = {};
    c = __builtin_amdgcn_wmma_f32_16x16x32_f16(
        /*neg_a=*/false, grayA, /*neg_b=*/false, ones,
        /*c_mod=*/(short)0, c, /*reuse_a=*/false, /*reuse_b=*/false);

    float t = ((c[0] + c[1]) + (c[2] + c[3])) + ((c[4] + c[5]) + (c[6] + c[7]));
    t += __shfl_xor(t, 16, 32);
    if ((tid & 31) == 0) atomicAdd(&gsum[b], t);
}

// Kernel B: in-place contrast / saturation / hue on the imgs region of d_out.
__global__ __launch_bounds__(256) void PairedAug_color_kernel(
    float* __restrict__ out_imgs, const float* __restrict__ gsum,
    const float* __restrict__ contrast, const float* __restrict__ saturation,
    const float* __restrict__ hue)
{
    const int idx = (int)blockIdx.x * 256 + (int)threadIdx.x;   // 0 .. B*HW-1
    const int b   = idx >> 18;                                  // HW = 2^18
    const int p   = idx & (HWn - 1);

    const float cf   = contrast[b];
    const float sf   = saturation[b];
    const float hfv  = hue[b];
    const float mean = gsum[b] * (1.0f / (float)HWn);

    float* pix = out_imgs + (size_t)(b * 3) * HWn + p;
    float r = pix[0];
    float g = pix[HWn];
    float bl = pix[2 * HWn];

    // contrast (mean of gray of brightness-clipped image)
    r  = clamp01(cf * r  + (1.0f - cf) * mean);
    g  = clamp01(cf * g  + (1.0f - cf) * mean);
    bl = clamp01(cf * bl + (1.0f - cf) * mean);

    // saturation
    const float gray = 0.299f * r + 0.587f * g + 0.114f * bl;
    r  = clamp01(sf * r  + (1.0f - sf) * gray);
    g  = clamp01(sf * g  + (1.0f - sf) * gray);
    bl = clamp01(sf * bl + (1.0f - sf) * gray);

    // rgb -> hsv
    const float maxc  = fmaxf(fmaxf(r, g), bl);
    const float minc  = fminf(fminf(r, g), bl);
    const float delta = maxc - minc;
    const float eps   = 1e-8f;
    const float s     = delta / (maxc + eps);
    const float dd    = delta + eps;
    const float rc    = (maxc - r)  / dd;
    const float gc    = (maxc - g)  / dd;
    const float bc    = (maxc - bl) / dd;
    float h6 = (maxc == r) ? (bc - gc)
             : ((maxc == g) ? (2.0f + rc - bc) : (4.0f + gc - rc));
    float h = h6 * (1.0f / 6.0f);
    h = h - floorf(h);
    h = h + hfv;
    h = h - floorf(h);

    // hsv -> rgb
    const float v  = maxc;
    const float f6 = h * 6.0f;
    const float fi = floorf(f6);
    const float f  = f6 - fi;
    const int   i  = ((int)fi) % 6;
    const float pp = v * (1.0f - s);
    const float q  = v * (1.0f - f * s);
    const float t  = v * (1.0f - (1.0f - f) * s);

    float ro, go, bo;
    if      (i == 0) { ro = v;  go = t;  bo = pp; }
    else if (i == 1) { ro = q;  go = v;  bo = pp; }
    else if (i == 2) { ro = pp; go = v;  bo = t;  }
    else if (i == 3) { ro = pp; go = q;  bo = v;  }
    else if (i == 4) { ro = t;  go = pp; bo = v;  }
    else             { ro = v;  go = pp; bo = q;  }

    pix[0]       = clamp01(ro);
    pix[HWn]     = clamp01(go);
    pix[2 * HWn] = clamp01(bo);
}

extern "C" void kernel_launch(void* const* d_in, const int* in_sizes, int n_in,
                              void* d_out, int out_size, void* d_ws, size_t ws_size,
                              hipStream_t stream) {
    (void)in_sizes; (void)n_in; (void)out_size; (void)ws_size;

    const float* imgs      = (const float*)d_in[0];
    const float* masks     = (const float*)d_in[1];
    const int*   hflip     = (const int*)  d_in[2];
    const int*   vflip     = (const int*)  d_in[3];
    const float* angles    = (const float*)d_in[4];
    const float* translate = (const float*)d_in[5];
    const float* scale     = (const float*)d_in[6];
    const float* bright    = (const float*)d_in[7];
    const float* contrast  = (const float*)d_in[8];
    const float* sat       = (const float*)d_in[9];
    const float* hue       = (const float*)d_in[10];

    float* out_imgs  = (float*)d_out;
    float* out_masks = out_imgs + (size_t)Bn * 3 * HWn;
    float* gsum      = (float*)d_ws;   // 32 floats

    PairedAug_zero_kernel<<<1, 32, 0, stream>>>(gsum);

    // 32 images * 64 blocks, 256 threads, 16 pixels/thread == exactly B*H*W pixels
    PairedAug_geom_kernel<<<Bn * 64, 256, 0, stream>>>(
        imgs, masks, hflip, vflip, angles, translate, scale, bright,
        out_imgs, out_masks, gsum);

    PairedAug_color_kernel<<<(Bn * HWn) / 256, 256, 0, stream>>>(
        out_imgs, gsum, contrast, sat, hue);
}
